// GQABlock_56564719289047
// MI455X (gfx1250) — matrix operand order repaired and checked by reading, not compiled, not run on hardware
//
#include <hip/hip_runtime.h>
#include <hip/hip_bf16.h>

#define S_LEN   2048
#define E_DIM   4096
#define HQ      32
#define HKV     8
#define D_HEAD  128
#define CACHE_T 1024
#define T_LEN   3072
#define QSCALE  0.08838834764831845f   // 1/sqrt(128)

typedef __attribute__((ext_vector_type(8)))  float  v8f;
typedef __attribute__((ext_vector_type(16))) __bf16 v16bf;
typedef __attribute__((ext_vector_type(8)))  __bf16 bf16x8;

union Frag16 { v16bf v; bf16x8 h[2]; };

__device__ __forceinline__ v8f wmma_bf16(v16bf a, v16bf b, v8f c) {
  // D = A(16x32) * B(32x16) + C, f32 accumulate
  return __builtin_amdgcn_wmma_f32_16x16x32_bf16(false, a, false, b, (short)0, c,
                                                 false, false);
}

// ---------------------------------------------------------------------------
// Copy KV cache (f32) into the K/V output regions of d_out.
// ---------------------------------------------------------------------------
__global__ __launch_bounds__(256) void copy_cache_kernel(
    const float* __restrict__ kc, const float* __restrict__ vc,
    float* __restrict__ Kout, float* __restrict__ Vout)
{
  int idx = blockIdx.x * 256 + threadIdx.x;          // over HKV*CACHE_T*D
  int h   = idx / (CACHE_T * D_HEAD);
  int rem = idx % (CACHE_T * D_HEAD);                // t*D + d
  size_t dst = (size_t)h * T_LEN * D_HEAD + rem;
  Kout[dst] = kc[idx];
  Vout[dst] = vc[idx];
}

// ---------------------------------------------------------------------------
// Convert concatenated K (f32, d_out) -> bf16 Kb[h][t][d] (coalesced).
// ---------------------------------------------------------------------------
__global__ __launch_bounds__(256) void convert_k_kernel(
    const float* __restrict__ Kf, __bf16* __restrict__ Kb)
{
  int idx = blockIdx.x * 256 + threadIdx.x;          // over HKV*T*D
  Kb[idx] = (__bf16)Kf[idx];
}

// ---------------------------------------------------------------------------
// Tiled transpose+convert: Vt[h][d][t] (bf16) = V (f32, d_out [h][t][d]).
// 64x64 tile per block through padded LDS; both global sides coalesced.
// ---------------------------------------------------------------------------
__global__ __launch_bounds__(256) void transpose_v_kernel(
    const float* __restrict__ Vf, __bf16* __restrict__ Vt)
{
  __shared__ __attribute__((aligned(16))) __bf16 tile[64][72];  // pad 72 vs 64
  const int hb    = blockIdx.z;
  const int tBase = blockIdx.x * 64;
  const int dBase = blockIdx.y * 64;
  const int tid   = threadIdx.x;
  const int r  = tid >> 2;            // 0..63
  const int cg = (tid & 3) << 4;      // 0,16,32,48

  const float4* src = (const float4*)(Vf + ((size_t)hb * T_LEN + tBase + r) * D_HEAD
                                      + dBase + cg);
  float4 f0 = src[0], f1 = src[1], f2 = src[2], f3 = src[3];
  float v[16] = {f0.x, f0.y, f0.z, f0.w, f1.x, f1.y, f1.z, f1.w,
                 f2.x, f2.y, f2.z, f2.w, f3.x, f3.y, f3.z, f3.w};
#pragma unroll
  for (int i = 0; i < 16; i++) tile[r][cg + i] = (__bf16)v[i];
  __syncthreads();
  __bf16* dst = Vt + ((size_t)hb * D_HEAD + dBase + r) * T_LEN + tBase + cg;
#pragma unroll
  for (int i = 0; i < 16; i++) dst[i] = tile[cg + i][r];
}

// ---------------------------------------------------------------------------
// Tiled GEMM: C[M,N] = A[M,4096] x B[4096,N], 128x128 block, 8 waves, LDS
// double-buffered; each wave computes 64x32 (4x2 WMMA subtiles).
// MODE 0: A=x f32, write Q bf16 head-major scaled by 1/sqrt(D)
// MODE 1: A=x f32, write K_new f32 into d_out K region (offset CACHE_T)
// MODE 2: A=x f32, write V_new f32 into d_out V region
// MODE 3: A=res bf16, write out f32 row-major
// ---------------------------------------------------------------------------
template<int MODE>
__global__ __launch_bounds__(256) void gemm_kernel(
    const float*  __restrict__ Af, const __bf16* __restrict__ Ab,
    const float*  __restrict__ Bw,
    float* __restrict__ outF, __bf16* __restrict__ outB, int N)
{
  __shared__ __attribute__((aligned(16))) __bf16 lA [2][128 * 32]; // [m][k]
  __shared__ __attribute__((aligned(16))) __bf16 lBt[2][128 * 32]; // [n][k]

  const int tid     = threadIdx.x;
  const int lane    = tid & 31;
  const int wave    = tid >> 5;
  const int wm      = wave >> 2;     // 0..1
  const int wn      = wave & 3;      // 0..3
  const int rowBase = blockIdx.y * 128;
  const int colBase = blockIdx.x * 128;
  const int n16     = lane & 15;
  const int halfsel = lane >> 4;

  // staging thread mapping
  const int rA  = tid >> 1;          // 0..127   (A tile row)
  const int cgA = (tid & 1) << 4;    // 0 or 16  (A tile col group)
  const int krB = tid >> 3;          // 0..31    (B tile row = k)
  const int cgB = (tid & 7) << 4;    // 0..112   (B tile col group)

  v8f acc[4][2];
#pragma unroll
  for (int i = 0; i < 4; i++)
#pragma unroll
    for (int j = 0; j < 2; j++) acc[i][j] = (v8f)0.0f;

  float4 aR[4];      // staged A tile regs (f32 modes)
  bf16x8 aB[2];      // staged A tile regs (bf16 mode)
  float4 bR[4];      // staged B tile regs

  // ---- issue global loads for tile at k-offset kk into registers ----
  auto issue_loads = [&](int kk) {
    if (MODE == 3) {
      const bf16x8* s = (const bf16x8*)(Ab + (size_t)(rowBase + rA) * E_DIM + kk + cgA);
      aB[0] = s[0]; aB[1] = s[1];
    } else {
      const float4* s = (const float4*)(Af + (size_t)(rowBase + rA) * E_DIM + kk + cgA);
      aR[0] = s[0]; aR[1] = s[1]; aR[2] = s[2]; aR[3] = s[3];
    }
    const float4* s = (const float4*)(Bw + (size_t)(kk + krB) * N + colBase + cgB);
    bR[0] = s[0]; bR[1] = s[1]; bR[2] = s[2]; bR[3] = s[3];
  };

  // ---- store staged registers into LDS buffer `buf` (bf16, B transposed) ----
  auto store_tile = [&](int buf) {
    if (MODE == 3) {
      *(bf16x8*)&lA[buf][rA * 32 + cgA]     = aB[0];
      *(bf16x8*)&lA[buf][rA * 32 + cgA + 8] = aB[1];
    } else {
      float v[16] = {aR[0].x, aR[0].y, aR[0].z, aR[0].w, aR[1].x, aR[1].y, aR[1].z, aR[1].w,
                     aR[2].x, aR[2].y, aR[2].z, aR[2].w, aR[3].x, aR[3].y, aR[3].z, aR[3].w};
      __bf16* dst = &lA[buf][rA * 32 + cgA];
#pragma unroll
      for (int i = 0; i < 16; i++) dst[i] = (__bf16)v[i];
    }
    float v[16] = {bR[0].x, bR[0].y, bR[0].z, bR[0].w, bR[1].x, bR[1].y, bR[1].z, bR[1].w,
                   bR[2].x, bR[2].y, bR[2].z, bR[2].w, bR[3].x, bR[3].y, bR[3].z, bR[3].w};
#pragma unroll
    for (int i = 0; i < 16; i++) lBt[buf][(cgB + i) * 32 + krB] = (__bf16)v[i];
  };

  issue_loads(0);
  int buf = 0;
  for (int kk = 0; kk < E_DIM; kk += 32) {
    store_tile(buf);
    __syncthreads();                       // single barrier per k-step
    if (kk + 32 < E_DIM) issue_loads(kk + 32);  // overlap next-tile loads w/ WMMA

    Frag16 bfrag[2];
#pragma unroll
    for (int ns = 0; ns < 2; ns++) {
      int n = wn * 32 + ns * 16 + n16;
      const bf16x8* p = (const bf16x8*)&lBt[buf][n * 32 + halfsel * 8];
      bfrag[ns].h[0] = p[0];
      bfrag[ns].h[1] = p[2];
    }
#pragma unroll
    for (int ms = 0; ms < 4; ms++) {
      int m = wm * 64 + ms * 16 + n16;
      Frag16 afrag;
      afrag.h[0] = *(const bf16x8*)&lA[buf][m * 32 + halfsel * 8];
      afrag.h[1] = *(const bf16x8*)&lA[buf][m * 32 + 16 + halfsel * 8];
#pragma unroll
      for (int ns = 0; ns < 2; ns++)
        acc[ms][ns] = wmma_bf16(afrag.v, bfrag[ns].v, acc[ms][ns]);
    }
    buf ^= 1;
  }

  // ---- epilogue: C layout row = j + halfsel*8, col = lane%16 ----
#pragma unroll
  for (int ms = 0; ms < 4; ms++) {
#pragma unroll
    for (int ns = 0; ns < 2; ns++) {
#pragma unroll
      for (int j = 0; j < 8; j++) {
        int row = rowBase + wm * 64 + ms * 16 + (halfsel << 3) + j;
        int col = colBase + wn * 32 + ns * 16 + n16;
        float v = acc[ms][ns][j];
        if (MODE == 0) {
          int head = col >> 7, d = col & 127;
          outB[((size_t)head * S_LEN + row) * D_HEAD + d] = (__bf16)(v * QSCALE);
        } else if (MODE == 1 || MODE == 2) {
          int head = col >> 7, d = col & 127;
          outF[((size_t)head * T_LEN + CACHE_T + row) * D_HEAD + d] = v;
        } else {
          outF[(size_t)row * E_DIM + col] = v;
        }
      }
    }
  }
}

// ---------------------------------------------------------------------------
// One 32-key flash-attention step for a 16-query tile. MASK only in the final
// tile (exactly one per wave given CACHE=1024 and qbase % 16 == 0).
// ---------------------------------------------------------------------------
__device__ __forceinline__ void attn_tile(
    int tb, bool domask, int qbase, int n16, int half,
    const Frag16 (&qf)[4], const __bf16* __restrict__ Kh,
    const __bf16* __restrict__ Vh, __bf16* __restrict__ lPw,
    v8f (&acc)[8], float (&mrow)[8], float (&lrow)[8])
{
  // ---- scores: two 16x16 tiles over 32 keys ----
  v8f s0 = (v8f)0.0f, s1 = (v8f)0.0f;
  {
    const __bf16* k0 = Kh + (size_t)(tb + n16) * D_HEAD + half * 16;
    const __bf16* k1 = Kh + (size_t)(tb + 16 + n16) * D_HEAD + half * 16;
#pragma unroll
    for (int c = 0; c < 4; c++) {
      Frag16 kf;
      kf.h[0] = *(const bf16x8*)(k0 + c * 32);
      kf.h[1] = *(const bf16x8*)(k0 + c * 32 + 8);
      s0 = wmma_bf16(qf[c].v, kf.v, s0);
      kf.h[0] = *(const bf16x8*)(k1 + c * 32);
      kf.h[1] = *(const bf16x8*)(k1 + c * 32 + 8);
      s1 = wmma_bf16(qf[c].v, kf.v, s1);
    }
  }
  // ---- (optional) causal+cache mask ----
  float p0[8], p1[8];
#pragma unroll
  for (int j = 0; j < 8; j++) {
    float a = s0[j], b = s1[j];
    if (domask) {
      int lim = qbase + half * 8 + j + CACHE_T;
      if (tb + n16 > lim)      a = -1e30f;
      if (tb + 16 + n16 > lim) b = -1e30f;
    }
    p0[j] = a; p1[j] = b;
  }
  // ---- online softmax (row r spread over 16 lanes of one half-wave) ----
#pragma unroll
  for (int j = 0; j < 8; j++) {
    float mx = fmaxf(p0[j], p1[j]);
    mx = fmaxf(mx, __shfl_xor(mx, 1));
    mx = fmaxf(mx, __shfl_xor(mx, 2));
    mx = fmaxf(mx, __shfl_xor(mx, 4));
    mx = fmaxf(mx, __shfl_xor(mx, 8));
    float mnew = fmaxf(mrow[j], mx);
    float corr = __expf(mrow[j] - mnew);
    mrow[j] = mnew;
    float e0 = __expf(p0[j] - mnew);
    float e1 = __expf(p1[j] - mnew);
    lrow[j] = lrow[j] * corr + e0 + e1;
    p0[j] = e0; p1[j] = e1;
#pragma unroll
    for (int dt = 0; dt < 8; dt++) acc[dt][j] = acc[dt][j] * corr;
  }
  // ---- P: C-layout -> A-layout via per-wave LDS scratch ----
#pragma unroll
  for (int j = 0; j < 8; j++) {
    lPw[(half * 8 + j) * 32 + n16]      = (__bf16)p0[j];
    lPw[(half * 8 + j) * 32 + 16 + n16] = (__bf16)p1[j];
  }
  Frag16 pf;
  pf.h[0] = *(const bf16x8*)&lPw[n16 * 32 + half * 8];
  pf.h[1] = *(const bf16x8*)&lPw[n16 * 32 + 16 + half * 8];
  // ---- O += P(16x32) x V(32x128), 8 d-tiles ----
#pragma unroll
  for (int dt = 0; dt < 8; dt++) {
    const __bf16* vp = Vh + (size_t)(dt * 16 + n16) * T_LEN + tb + half * 8;
    Frag16 vf;
    vf.h[0] = *(const bf16x8*)(vp);
    vf.h[1] = *(const bf16x8*)(vp + 16);
    acc[dt] = wmma_bf16(pf.v, vf.v, acc[dt]);
  }
}

// ---------------------------------------------------------------------------
// Flash attention. 1 wave = 16 queries of one head; 4 waves/WG.
// Query head h uses KV head h % 8 (reference reshape (RATIO, HKV)).
// Q pre-scaled by 1/sqrt(D). K+V bf16 (12MB total) are L2-resident (192MB).
// ---------------------------------------------------------------------------
__global__ __launch_bounds__(128) void attn_kernel(
    const __bf16* __restrict__ Qb, const __bf16* __restrict__ Kb,
    const __bf16* __restrict__ Vt, __bf16* __restrict__ resb)
{
  __shared__ __attribute__((aligned(16))) __bf16 lP[4][16 * 32];

  const int tid   = threadIdx.x;
  const int lane  = tid & 31;
  const int wave  = tid >> 5;
  const int h     = blockIdx.y;
  const int kv    = h & (HKV - 1);
  const int qbase = blockIdx.x * 64 + wave * 16;
  const int n16   = lane & 15;
  const int half  = lane >> 4;

  // Q fragments (A operand), 4 chunks of K=32 over D=128
  Frag16 qf[4];
  {
    const __bf16* qrow = Qb + ((size_t)h * S_LEN + qbase + n16) * D_HEAD;
#pragma unroll
    for (int c = 0; c < 4; c++) {
      qf[c].h[0] = *(const bf16x8*)(qrow + c * 32 + half * 16);
      qf[c].h[1] = *(const bf16x8*)(qrow + c * 32 + half * 16 + 8);
    }
  }

  v8f acc[8];
#pragma unroll
  for (int i = 0; i < 8; i++) acc[i] = (v8f)0.0f;
  float mrow[8], lrow[8];
#pragma unroll
  for (int j = 0; j < 8; j++) { mrow[j] = -1e30f; lrow[j] = 0.0f; }

  const __bf16* Kh = Kb + (size_t)kv * T_LEN * D_HEAD;
  const __bf16* Vh = Vt + (size_t)kv * D_HEAD * T_LEN;
  __bf16* lPw = &lP[wave][0];

  const int tEnd   = qbase + CACHE_T + 16;            // exclusive key bound
  const int tbLast = (((tEnd + 31) >> 5) - 1) << 5;   // only tile needing mask

  for (int tb = 0; tb < tbLast; tb += 32)
    attn_tile(tb, false, qbase, n16, half, qf, Kh, Vh, lPw, acc, mrow, lrow);
  attn_tile(tbLast, true, qbase, n16, half, qf, Kh, Vh, lPw, acc, mrow, lrow);

  // ---- finalize: divide by row sums, write res bf16 [s][h*D + d] ----
#pragma unroll
  for (int j = 0; j < 8; j++) {
    float l = lrow[j];
    l += __shfl_xor(l, 1);
    l += __shfl_xor(l, 2);
    l += __shfl_xor(l, 4);
    l += __shfl_xor(l, 8);
    lrow[j] = 1.0f / l;
  }
#pragma unroll
  for (int dt = 0; dt < 8; dt++) {
#pragma unroll
    for (int j = 0; j < 8; j++) {
      int srow = qbase + half * 8 + j;
      resb[(size_t)srow * E_DIM + h * D_HEAD + dt * 16 + n16] =
          (__bf16)(acc[dt][j] * lrow[j]);
    }
  }
}

// ---------------------------------------------------------------------------
extern "C" void kernel_launch(void* const* d_in, const int* in_sizes, int n_in,
                              void* d_out, int out_size, void* d_ws, size_t ws_size,
                              hipStream_t stream) {
  (void)in_sizes; (void)n_in; (void)out_size; (void)ws_size;
  const float* x   = (const float*)d_in[0];
  const float* w_q = (const float*)d_in[1];
  const float* w_k = (const float*)d_in[2];
  const float* w_v = (const float*)d_in[3];
  const float* w_o = (const float*)d_in[4];
  const float* kc  = (const float*)d_in[5];
  const float* vc  = (const float*)d_in[6];

  float* out  = (float*)d_out;
  float* Kout = out  + (size_t)S_LEN * E_DIM;          // [HKV][T][D] f32
  float* Vout = Kout + (size_t)HKV * T_LEN * D_HEAD;   // [HKV][T][D] f32

  __bf16* Qb   = (__bf16*)d_ws;                        // [HQ][S][D]   16 MB
  __bf16* resb = Qb   + (size_t)HQ  * S_LEN * D_HEAD;  // [S][E]       16 MB
  __bf16* Kb   = resb + (size_t)S_LEN * E_DIM;         // [HKV][T][D]   6 MB
  __bf16* Vt   = Kb   + (size_t)HKV * T_LEN * D_HEAD;  // [HKV][D][T]   6 MB

  copy_cache_kernel<<<(HKV * CACHE_T * D_HEAD) / 256, 256, 0, stream>>>(kc, vc, Kout, Vout);

  gemm_kernel<0><<<dim3(E_DIM / 128, S_LEN / 128), 256, 0, stream>>>(
      x, nullptr, w_q, nullptr, Qb, E_DIM);
  gemm_kernel<1><<<dim3((HKV * D_HEAD) / 128, S_LEN / 128), 256, 0, stream>>>(
      x, nullptr, w_k, Kout, nullptr, HKV * D_HEAD);
  gemm_kernel<2><<<dim3((HKV * D_HEAD) / 128, S_LEN / 128), 256, 0, stream>>>(
      x, nullptr, w_v, Vout, nullptr, HKV * D_HEAD);

  convert_k_kernel<<<(HKV * T_LEN * D_HEAD) / 256, 256, 0, stream>>>(Kout, Kb);
  transpose_v_kernel<<<dim3(T_LEN / 64, D_HEAD / 64, HKV), 256, 0, stream>>>(Vout, Vt);

  attn_kernel<<<dim3(S_LEN / 64, HQ), 128, 0, stream>>>(Qb, Kb, Vt, resb);

  gemm_kernel<3><<<dim3(E_DIM / 128, S_LEN / 128), 256, 0, stream>>>(
      nullptr, resb, w_o, out, nullptr, E_DIM);
}
